// CausalSelfAttn_21053929685056
// MI455X (gfx1250) — matrix operand order repaired and checked by reading, not compiled, not run on hardware
//
#include <hip/hip_runtime.h>
#include <hip/hip_bf16.h>

typedef __attribute__((ext_vector_type(16))) __bf16 v16bf;
typedef __attribute__((ext_vector_type(8)))  float  v8f;

union Frag { v16bf v; uint4 q[2]; };

#define S_LEN 4096
#define D_MODEL 1024
#define N_HEADS 16
#define D_K 64
#define ROPE_THETA 10000.0f

__device__ __forceinline__ unsigned short f2bf(float f) {
    unsigned int u = __float_as_uint(f);
    u += 0x7fffu + ((u >> 16) & 1u);   // round-to-nearest-even
    return (unsigned short)(u >> 16);
}

// Async 16-byte global -> LDS copy (gfx1250 ASYNCcnt path).
__device__ __forceinline__ void async_cp16(const unsigned short* g,
                                           unsigned short* l) {
    unsigned int laddr = (unsigned int)(unsigned long long)l;
    asm volatile("global_load_async_to_lds_b128 %0, %1, off"
                 :: "v"(laddr), "v"(g) : "memory");
}
__device__ __forceinline__ void async_wait0() {
    asm volatile("s_wait_asynccnt 0" ::: "memory");
}

// ---------------- Kernel 0: fp32 -> bf16 convert ----------------
__global__ void cvt_bf16_kernel(const float* __restrict__ src,
                                unsigned short* __restrict__ dst, int n) {
    int i = blockIdx.x * blockDim.x + threadIdx.x;
    if (i < n) dst[i] = f2bf(src[i]);
}

// ---------------- Kernel 1: QKV projection + RoPE ----------------
// Wave computes a 32x64 tile (two 16-row halves share every B fragment).
__global__ __launch_bounds__(256) void qkv_rope_kernel(
    const unsigned short* __restrict__ xb,   // [S][D] bf16
    const unsigned short* __restrict__ wqb,  // [D][D] bf16 (row = out dim)
    const unsigned short* __restrict__ wkb,
    const unsigned short* __restrict__ wvb,
    const float* __restrict__ bq, const float* __restrict__ bk,
    const float* __restrict__ bv,
    unsigned short* __restrict__ qout,       // [H][S][DK] bf16 (RoPE'd, *1/8)
    unsigned short* __restrict__ kout,       // [H][S][DK] bf16 (RoPE'd)
    unsigned short* __restrict__ vtout)      // [H][DK][S] bf16 (transposed)
{
    const int lane = threadIdx.x & 31;
    const int wid  = (blockIdx.x * 256 + threadIdx.x) >> 5;
    const int mat  = wid / (128 * N_HEADS);          // 0=q 1=k 2=v
    const int rem  = wid % (128 * N_HEADS);
    const int sb   = rem >> 4;                       // 0..127 (32-row tile)
    const int h    = rem & 15;
    const int hh   = lane >> 4;
    const int ln   = lane & 15;

    const unsigned short* W = (mat == 0) ? wqb : (mat == 1) ? wkb : wvb;
    const float* bias       = (mat == 0) ? bq  : (mat == 1) ? bk  : bv;

    v8f acc[2][4] = {};

    for (int c0 = 0; c0 < D_MODEL; c0 += 32) {
        Frag a[2];
#pragma unroll
        for (int r = 0; r < 2; ++r) {
            const unsigned short* ap =
                xb + (size_t)(sb * 32 + 16 * r + ln) * D_MODEL + c0;
            a[r].q[0] = *(const uint4*)(ap + 16 * hh);
            a[r].q[1] = *(const uint4*)(ap + 16 * hh + 8);
        }
#pragma unroll
        for (int t = 0; t < 4; ++t) {
            Frag b;
            const unsigned short* wp =
                W + (size_t)(h * D_K + 16 * t + ln) * D_MODEL + c0 + 16 * hh;
            b.q[0] = *(const uint4*)(wp);
            b.q[1] = *(const uint4*)(wp + 8);
#pragma unroll
            for (int r = 0; r < 2; ++r)
                acc[r][t] = __builtin_amdgcn_wmma_f32_16x16x32_bf16(
                    false, a[r].v, false, b.v, (short)0, acc[r][t],
                    false, false);
        }
    }

#pragma unroll
    for (int r = 0; r < 2; ++r) {
        const int s_base = sb * 32 + 16 * r + 8 * hh;
#pragma unroll
        for (int t = 0; t < 4; ++t) {
            const int col = 16 * t + ln;
            const float bsc = bias[h * D_K + col];
            float vals[8];
#pragma unroll
            for (int i = 0; i < 8; ++i) vals[i] = acc[r][t][i] + bsc;

            if (mat < 2) {
                const float invf =
                    __powf(ROPE_THETA, -(float)(col & ~1) / (float)D_K);
                const bool oddc = (col & 1);
#pragma unroll
                for (int i = 0; i < 8; ++i) {
                    float other = __shfl_xor(vals[i], 1, 32);
                    float ang = (float)(s_base + i) * invf;
                    float sn, cs;
                    __sincosf(ang, &sn, &cs);
                    vals[i] = oddc ? (other * sn + vals[i] * cs)
                                   : (vals[i] * cs - other * sn);
                    if (mat == 0) vals[i] *= 0.125f;
                }
                unsigned short* outp = (mat == 0) ? qout : kout;
#pragma unroll
                for (int i = 0; i < 8; ++i)
                    outp[((size_t)h * S_LEN + s_base + i) * D_K + col] =
                        f2bf(vals[i]);
            } else {
#pragma unroll
                for (int i = 0; i < 8; i += 2) {
                    unsigned int p = (unsigned int)f2bf(vals[i]) |
                                     ((unsigned int)f2bf(vals[i + 1]) << 16);
                    *(unsigned int*)(vtout + ((size_t)h * D_K + col) * S_LEN +
                                     s_base + i) = p;
                }
            }
        }
    }
}

// ------- Kernel 2: flash attention, cooperative async K/V staging ---------
__global__ __launch_bounds__(256) void attn_kernel(
    const unsigned short* __restrict__ qws,  // [H][S][DK]
    const unsigned short* __restrict__ kws,  // [H][S][DK]
    const unsigned short* __restrict__ vtws, // [H][DK][S]
    unsigned short* __restrict__ cout)       // [S][D] concat bf16
{
    __shared__ unsigned short kbuf[2][32 * 64];   // K block  [32 k][64 d]
    __shared__ unsigned short vbuf[2][64 * 32];   // V block  [64 d][32 k]
    __shared__ unsigned short plds[8 * 16 * 32];  // per-wave P scratch

    const int tid  = threadIdx.x;
    const int lane = tid & 31;
    const int wib  = tid >> 5;
    const int h    = blockIdx.x >> 5;         // 32 blocks per head
    const int qt0  = (blockIdx.x & 31) * 8;   // first q-tile of block
    const int qb   = qt0 + wib;               // this wave's q-tile
    const int hh   = lane >> 4;
    const int ln   = lane & 15;
    unsigned short* myp = plds + wib * 512;

    const int nkb   = (qb + 2) >> 1;          // my causal K-block count
    const int nkbmx = (qt0 + 7 + 2) >> 1;     // block-wide max

    auto issue_stage = [&](int kb, int buf) {
        const int k0 = kb * 32;
        const unsigned short* gk =
            kws + ((size_t)h * S_LEN + k0 + (tid >> 3)) * D_K + (tid & 7) * 8;
        const unsigned short* gv =
            vtws + ((size_t)h * D_K + (tid >> 2)) * S_LEN + k0 + (tid & 3) * 8;
        async_cp16(gk, &kbuf[buf][(tid >> 3) * 64 + (tid & 7) * 8]);
        async_cp16(gv, &vbuf[buf][(tid >> 2) * 32 + (tid & 3) * 8]);
    };

    Frag qa[2];
    const unsigned short* qrow = qws + ((size_t)h * S_LEN + qb * 16 + ln) * D_K;
#pragma unroll
    for (int c = 0; c < 2; ++c) {
        qa[c].q[0] = *(const uint4*)(qrow + 32 * c + 16 * hh);
        qa[c].q[1] = *(const uint4*)(qrow + 32 * c + 16 * hh + 8);
    }

    float m[8], l[8];
    v8f o[4] = {};
#pragma unroll
    for (int i = 0; i < 8; ++i) { m[i] = -3.0e38f; l[i] = 0.0f; }

    issue_stage(0, 0);

    for (int kb = 0; kb < nkbmx; ++kb) {
        async_wait0();        // my stage-kb async copies have landed in LDS
        __syncthreads();      // everyone's copies landed; prev compute done
        if (kb + 1 < nkbmx) issue_stage(kb + 1, (kb + 1) & 1);

        if (kb < nkb) {       // wave-uniform: EXEC stays all-ones for WMMA
            const unsigned short* kl = kbuf[kb & 1];
            const unsigned short* vl = vbuf[kb & 1];
            const int k0 = kb * 32;

            v8f sc[2] = {};
#pragma unroll
            for (int t = 0; t < 2; ++t) {
                const unsigned short* kr = kl + (16 * t + ln) * 64 + 16 * hh;
#pragma unroll
                for (int c = 0; c < 2; ++c) {
                    Frag b;
                    b.q[0] = *(const uint4*)(kr + 32 * c);
                    b.q[1] = *(const uint4*)(kr + 32 * c + 8);
                    sc[t] = __builtin_amdgcn_wmma_f32_16x16x32_bf16(
                        false, qa[c].v, false, b.v, (short)0, sc[t],
                        false, false);
                }
            }
            const int qg = qb * 16 + 8 * hh;
#pragma unroll
            for (int t = 0; t < 2; ++t) {
                const int kgl = k0 + 16 * t + ln;
#pragma unroll
                for (int i = 0; i < 8; ++i)
                    if (kgl > qg + i) sc[t][i] = -3.0e38f;
            }
            float alpha[8];
#pragma unroll
            for (int i = 0; i < 8; ++i) {
                float mx = fmaxf(sc[0][i], sc[1][i]);
                mx = fmaxf(mx, __shfl_xor(mx, 1, 32));
                mx = fmaxf(mx, __shfl_xor(mx, 2, 32));
                mx = fmaxf(mx, __shfl_xor(mx, 4, 32));
                mx = fmaxf(mx, __shfl_xor(mx, 8, 32));
                float mn = fmaxf(m[i], mx);
                alpha[i] = __expf(m[i] - mn);
                m[i] = mn;
                float p0 = __expf(sc[0][i] - mn);
                float p1 = __expf(sc[1][i] - mn);
                sc[0][i] = p0; sc[1][i] = p1;
                float rs = p0 + p1;
                rs += __shfl_xor(rs, 1, 32);
                rs += __shfl_xor(rs, 2, 32);
                rs += __shfl_xor(rs, 4, 32);
                rs += __shfl_xor(rs, 8, 32);
                l[i] = l[i] * alpha[i] + rs;
            }
#pragma unroll
            for (int t = 0; t < 4; ++t)
#pragma unroll
                for (int i = 0; i < 8; ++i) o[t][i] *= alpha[i];
#pragma unroll
            for (int t = 0; t < 2; ++t)
#pragma unroll
                for (int i = 0; i < 8; ++i)
                    myp[(i + 8 * hh) * 32 + 16 * t + ln] = f2bf(sc[t][i]);
            asm volatile("s_wait_dscnt 0" ::: "memory");
            Frag pa;
            pa.q[0] = *(const uint4*)(myp + ln * 32 + 16 * hh);
            pa.q[1] = *(const uint4*)(myp + ln * 32 + 16 * hh + 8);
#pragma unroll
            for (int t = 0; t < 4; ++t) {
                const unsigned short* vr = vl + (16 * t + ln) * 32 + 16 * hh;
                Frag b;
                b.q[0] = *(const uint4*)(vr);
                b.q[1] = *(const uint4*)(vr + 8);
                o[t] = __builtin_amdgcn_wmma_f32_16x16x32_bf16(
                    false, pa.v, false, b.v, (short)0, o[t], false, false);
            }
        }
    }
#pragma unroll
    for (int i = 0; i < 8; ++i) l[i] = 1.0f / l[i];
#pragma unroll
    for (int t = 0; t < 4; ++t) {
        const int col = h * D_K + 16 * t + ln;
#pragma unroll
        for (int i = 0; i < 8; ++i) {
            const int srow = qb * 16 + 8 * hh + i;
            cout[(size_t)srow * D_MODEL + col] = f2bf(o[t][i] * l[i]);
        }
    }
}

// ---------------- Kernel 3: output projection y = concat @ Wo^T + bo -------
// Wave computes a 32x64 tile (two 16-row halves share every B fragment).
__global__ __launch_bounds__(256) void out_proj_kernel(
    const unsigned short* __restrict__ cb,   // [S][D] bf16
    const unsigned short* __restrict__ wob,  // [D][D] bf16
    const float* __restrict__ bo,
    float* __restrict__ y)                   // [S][D] fp32
{
    const int lane = threadIdx.x & 31;
    const int wid  = (blockIdx.x * 256 + threadIdx.x) >> 5;
    const int sb   = wid >> 4;               // 0..127 (32-row tile)
    const int dt   = wid & 15;
    const int d0   = dt * 64;
    const int hh   = lane >> 4;
    const int ln   = lane & 15;

    v8f acc[2][4] = {};
    for (int c0 = 0; c0 < D_MODEL; c0 += 32) {
        Frag a[2];
#pragma unroll
        for (int r = 0; r < 2; ++r) {
            const unsigned short* ap =
                cb + (size_t)(sb * 32 + 16 * r + ln) * D_MODEL + c0;
            a[r].q[0] = *(const uint4*)(ap + 16 * hh);
            a[r].q[1] = *(const uint4*)(ap + 16 * hh + 8);
        }
#pragma unroll
        for (int t = 0; t < 4; ++t) {
            Frag b;
            const unsigned short* wp =
                wob + (size_t)(d0 + 16 * t + ln) * D_MODEL + c0 + 16 * hh;
            b.q[0] = *(const uint4*)(wp);
            b.q[1] = *(const uint4*)(wp + 8);
#pragma unroll
            for (int r = 0; r < 2; ++r)
                acc[r][t] = __builtin_amdgcn_wmma_f32_16x16x32_bf16(
                    false, a[r].v, false, b.v, (short)0, acc[r][t],
                    false, false);
        }
    }
#pragma unroll
    for (int r = 0; r < 2; ++r)
#pragma unroll
        for (int t = 0; t < 4; ++t) {
            const int col = d0 + 16 * t + ln;
            const float bb = bo[col];
#pragma unroll
            for (int i = 0; i < 8; ++i) {
                const int srow = sb * 32 + 16 * r + 8 * hh + i;
                y[(size_t)srow * D_MODEL + col] = acc[r][t][i] + bb;
            }
        }
}

extern "C" void kernel_launch(void* const* d_in, const int* in_sizes, int n_in,
                              void* d_out, int out_size, void* d_ws,
                              size_t ws_size, hipStream_t stream) {
    const float* x  = (const float*)d_in[0];
    const float* Wq = (const float*)d_in[1];
    const float* bq = (const float*)d_in[2];
    const float* Wk = (const float*)d_in[3];
    const float* bk = (const float*)d_in[4];
    const float* Wv = (const float*)d_in[5];
    const float* bv = (const float*)d_in[6];
    const float* Wo = (const float*)d_in[7];
    const float* bo = (const float*)d_in[8];
    float* y = (float*)d_out;

    const size_t MB = 1024 * 1024;
    unsigned char* ws = (unsigned char*)d_ws;
    unsigned short* xb   = (unsigned short*)(ws);              //  8 MiB
    unsigned short* wqb  = (unsigned short*)(ws + 8  * MB);    //  2 MiB
    unsigned short* wkb  = (unsigned short*)(ws + 10 * MB);    //  2 MiB
    unsigned short* wvb  = (unsigned short*)(ws + 12 * MB);    //  2 MiB
    unsigned short* wob  = (unsigned short*)(ws + 14 * MB);    //  2 MiB
    unsigned short* qws  = (unsigned short*)(ws + 16 * MB);    //  8 MiB
    unsigned short* kws  = (unsigned short*)(ws + 24 * MB);    //  8 MiB
    unsigned short* vtws = (unsigned short*)(ws + 32 * MB);    //  8 MiB
    unsigned short* cws  = (unsigned short*)(ws + 40 * MB);    //  8 MiB

    const int nx = S_LEN * D_MODEL;
    const int nw = D_MODEL * D_MODEL;
    cvt_bf16_kernel<<<(nx + 255) / 256, 256, 0, stream>>>(x, xb, nx);
    cvt_bf16_kernel<<<(nw + 255) / 256, 256, 0, stream>>>(Wq, wqb, nw);
    cvt_bf16_kernel<<<(nw + 255) / 256, 256, 0, stream>>>(Wk, wkb, nw);
    cvt_bf16_kernel<<<(nw + 255) / 256, 256, 0, stream>>>(Wv, wvb, nw);
    cvt_bf16_kernel<<<(nw + 255) / 256, 256, 0, stream>>>(Wo, wob, nw);

    // 3 mats * 16 heads * 128 32-row tiles = 6144 waves / 8 per block
    qkv_rope_kernel<<<768, 256, 0, stream>>>(xb, wqb, wkb, wvb, bq, bk, bv,
                                             qws, kws, vtws);
    // 16 heads * 256 q-tiles = 4096 waves / 8 per block
    attn_kernel<<<512, 256, 0, stream>>>(qws, kws, vtws, cws);
    // 128 32-row tiles * 16 d-tiles = 2048 waves / 8 per block
    out_proj_kernel<<<256, 256, 0, stream>>>(cws, wob, bo, y);
}
